// HierarchicalGNN_59983513256104
// MI455X (gfx1250) — matrix operand order repaired and checked by reading, not compiled, not run on hardware
//
#include <hip/hip_runtime.h>
#include <math.h>

typedef __attribute__((ext_vector_type(16))) _Float16 v16h;
typedef __attribute__((ext_vector_type(8)))  float    v8f;
typedef __attribute__((ext_vector_type(4)))  int      v4i;

#define HD 128           // hidden dim
#define NDIM 4           // x_local feature dim
#define FGD 7            // global feat dim
#define HC 64            // classifier hidden
#define NCLS 2
#define GEMM_THREADS 64  // 2 waves per block

static __device__ __forceinline__ void atomicMaxF(float* addr, float v) {
    // monotone-bits trick: works with -inf init, mixed signs
    if (v >= 0.0f) atomicMax((int*)addr, __float_as_int(v));
    else           atomicMin((unsigned int*)addr, (unsigned int)__float_as_int(v));
}

// ---------------- generic fills / converts ----------------
__global__ void k_fill(float* p, float v, long long n) {
    long long i = (long long)blockIdx.x * blockDim.x + threadIdx.x;
    if (i < n) p[i] = v;
}
__global__ void k_cvt_f16(const float* __restrict__ s, _Float16* __restrict__ d, long long n) {
    long long i = (long long)blockIdx.x * blockDim.x + threadIdx.x;
    if (i < n) d[i] = (_Float16)s[i];
}

// ---------------- input projection: h = x @ Win + bin ----------------
__global__ void k_in_proj(const float* __restrict__ x, const float* __restrict__ W,
                          const float* __restrict__ b, float* __restrict__ h, int T) {
    long long i = (long long)blockIdx.x * blockDim.x + threadIdx.x;
    if (i >= (long long)T * HD) return;
    int t = (int)(i >> 7), c = (int)(i & 127);
    const float* xr = x + (long long)t * NDIM;
    float acc = b[c];
#pragma unroll
    for (int k = 0; k < NDIM; ++k) acc += xr[k] * W[k * HD + c];
    h[i] = acc;
}

// ---------------- WMMA GEMM: C[n,128] = A[n,128] @ B[128,128] ----------------
// Block = 64 threads (2 waves) = one 16-row tile; wave w owns columns [64w, 64w+64)
// as 4 accumulators -> 16 v_wmma per wave, A tile fetched once per K-step.
// W is staged to LDS once per block via async load-to-LDS (ASYNCcnt path).
// A layout (16-bit 16x32): lane L holds M=L%16; half j -> K = (j<8?j:j+8) + (L>=16?8:0)
// B layout (32x16): lane L holds N=L%16; half j -> K = (j<8?j:j+8) + (L>=16?8:0)
// C layout: VGPR r, lane L -> M = r + 8*(L>=16), N = L%16
__global__ void __launch_bounds__(GEMM_THREADS)
k_wmma_gemm(const _Float16* __restrict__ A, const _Float16* __restrict__ Bm,
            float* __restrict__ C, int n) {
    __shared__ _Float16 Bs[HD * HD];     // 32 KB
    const int tid = threadIdx.x;

    // ---- stage full W (128x128 f16) into LDS ----
#if __has_builtin(__builtin_amdgcn_global_load_async_to_lds_b128)
    {
        typedef __attribute__((address_space(1))) v4i gv4i;
        typedef __attribute__((address_space(3))) v4i lv4i;
#pragma unroll
        for (int it = 0; it < (HD * HD) / (GEMM_THREADS * 8); ++it) {
            int idx = (it * GEMM_THREADS + tid) * 8;     // in halves (16B chunks)
            __builtin_amdgcn_global_load_async_to_lds_b128(
                (gv4i*)(Bm + idx), (lv4i*)(Bs + idx), 0, 0);
        }
#if __has_builtin(__builtin_amdgcn_s_wait_asynccnt)
        __builtin_amdgcn_s_wait_asynccnt(0);
#else
        asm volatile("s_wait_asynccnt 0x0" ::: "memory");
#endif
    }
#else
#pragma unroll
    for (int it = 0; it < (HD * HD) / (GEMM_THREADS * 8); ++it) {
        int idx = (it * GEMM_THREADS + tid) * 8;
        *(float4*)(Bs + idx) = *(const float4*)(Bm + idx);
    }
#endif
    __syncthreads();

    const int lane = tid & 31;
    const int wv   = tid >> 5;           // 0..1 -> 64-column half
    const int hi   = lane >> 4;
    const int rowb = blockIdx.x * 16;
    const _Float16* arow = A + (long long)(rowb + (lane & 15)) * HD;
    v8f c0 = {}, c1 = {}, c2 = {}, c3 = {};
#pragma unroll
    for (int k0 = 0; k0 < HD; k0 += 32) {
        v16h a;
        const _Float16* ap = arow + k0 + hi * 8;
#pragma unroll
        for (int j = 0; j < 8; ++j)  a[j] = ap[j];
#pragma unroll
        for (int j = 8; j < 16; ++j) a[j] = ap[j + 8];
        const _Float16* bp = Bs + (k0 + hi * 8) * HD + wv * 64 + (lane & 15);
        v16h b0, b1, b2, b3;
#pragma unroll
        for (int j = 0; j < 16; ++j) {
            const int ko = (j < 8 ? j : j + 8) * HD;
            b0[j] = bp[ko]; b1[j] = bp[ko + 16]; b2[j] = bp[ko + 32]; b3[j] = bp[ko + 48];
        }
        c0 = __builtin_amdgcn_wmma_f32_16x16x32_f16(false, a, false, b0, (short)0, c0, false, false);
        c1 = __builtin_amdgcn_wmma_f32_16x16x32_f16(false, a, false, b1, (short)0, c1, false, false);
        c2 = __builtin_amdgcn_wmma_f32_16x16x32_f16(false, a, false, b2, (short)0, c2, false, false);
        c3 = __builtin_amdgcn_wmma_f32_16x16x32_f16(false, a, false, b3, (short)0, c3, false, false);
    }
    const int mbase = rowb + hi * 8;
    const int nb    = wv * 64 + (lane & 15);
#pragma unroll
    for (int r = 0; r < 8; ++r) {
        float* cr = C + (long long)(mbase + r) * HD + nb;
        cr[0] = c0[r]; cr[16] = c1[r]; cr[32] = c2[r]; cr[48] = c3[r];
    }
}

// ---------------- per-node attention scores ----------------
__global__ void k_node_scores(const float* __restrict__ z, const float* __restrict__ asrc,
                              const float* __restrict__ adst, float* __restrict__ ss,
                              float* __restrict__ sd, int n, int nh) {
    long long i = (long long)blockIdx.x * blockDim.x + threadIdx.x;
    if (i >= (long long)n * nh) return;
    int node = (int)(i / nh), hh = (int)(i % nh);
    int d = HD / nh;
    const float* zr = z + (long long)node * HD + hh * d;
    const float* as = asrc + hh * d;
    const float* ad = adst + hh * d;
    float s1 = 0.f, s2 = 0.f;
    for (int k = 0; k < d; ++k) { float v = zr[k]; s1 += v * as[k]; s2 += v * ad[k]; }
    ss[i] = s1; sd[i] = s2;
}

// ---------------- edge-attr attention coefficient (per head) ----------------
__global__ void k_ec(const float* __restrict__ eW, const float* __restrict__ ae,
                     float* __restrict__ ec, int nh) {
    int hh = threadIdx.x;
    if (hh >= nh) return;
    int d = HD / nh;
    float s = 0.f;
    for (int k = 0; k < d; ++k) s += eW[hh * d + k] * ae[hh * d + k];
    ec[hh] = s;
}

// ---------------- edge score + LeakyReLU + segment max ----------------
__global__ void k_edge_score(const int* __restrict__ src, const int* __restrict__ dst,
                             const float* __restrict__ ss, const float* __restrict__ sd,
                             const float* __restrict__ eattr, const float* __restrict__ ec,
                             float* __restrict__ score, float* __restrict__ m,
                             long long E, int nh) {
    long long i = (long long)blockIdx.x * blockDim.x + threadIdx.x;
    if (i >= E * nh) return;
    long long e = i / nh; int hh = (int)(i % nh);
    float s = ss[(long long)src[e] * nh + hh] + sd[(long long)dst[e] * nh + hh];
    if (eattr) s += eattr[e] * ec[hh];
    s = s > 0.f ? s : 0.2f * s;
    score[i] = s;
    atomicMaxF(&m[(long long)dst[e] * nh + hh], s);
}

__global__ void k_m_fix(float* __restrict__ m, long long n) {
    long long i = (long long)blockIdx.x * blockDim.x + threadIdx.x;
    if (i >= n) return;
    float v = m[i];
    if (!isfinite(v)) m[i] = 0.f;
}

// ---------------- edge exp + segment sum ----------------
__global__ void k_edge_exp(const int* __restrict__ dst, const float* __restrict__ m,
                           float* __restrict__ score, float* __restrict__ den,
                           long long E, int nh) {
    long long i = (long long)blockIdx.x * blockDim.x + threadIdx.x;
    if (i >= E * nh) return;
    long long e = i / nh; int hh = (int)(i % nh);
    float ex = expf(score[i] - m[(long long)dst[e] * nh + hh]);
    score[i] = ex;
    atomicAdd(&den[(long long)dst[e] * nh + hh], ex);
}

// ---------------- weighted scatter aggregation (4 channels / thread) ----------------
__global__ void k_edge_agg(const int* __restrict__ src, const int* __restrict__ dst,
                           const float* __restrict__ z, const float* __restrict__ ex,
                           const float* __restrict__ den, float* __restrict__ agg,
                           long long E, int nh) {
    long long i = (long long)blockIdx.x * blockDim.x + threadIdx.x;
    if (i >= E * 32) return;
    long long e = i >> 5;
    int c4 = (int)(i & 31) * 4;            // 4 consecutive channels, same head (d%4==0)
    int d = HD / nh;
    int hh = c4 / d;
    int dn = dst[e], sn = src[e];
    float alpha = ex[e * nh + hh] / fmaxf(den[(long long)dn * nh + hh], 1e-16f);
    const float4 zv = *(const float4*)(z + (long long)sn * HD + c4);
    float* ap = agg + (long long)dn * HD + c4;
    atomicAdd(ap + 0, alpha * zv.x);
    atomicAdd(ap + 1, alpha * zv.y);
    atomicAdd(ap + 2, alpha * zv.z);
    atomicAdd(ap + 3, alpha * zv.w);
}

// ---------------- residual + bias + ELU (in-place on x) ----------------
__global__ void k_res_elu(float* __restrict__ x, const float* __restrict__ agg,
                          const float* __restrict__ b, long long rows) {
    long long i = (long long)blockIdx.x * blockDim.x + threadIdx.x;
    if (i >= rows * HD) return;
    int c = (int)(i & 127);
    float v = x[i] + agg[i] + b[c];
    x[i] = v > 0.f ? v : (expf(v) - 1.f);
}

// ---------------- mean pooling ----------------
__global__ void k_pool_sum(const float* __restrict__ h, const int* __restrict__ nb,
                           float* __restrict__ emb, int T) {
    long long i = (long long)blockIdx.x * blockDim.x + threadIdx.x;
    if (i >= (long long)T * 32) return;
    int t = (int)(i >> 5), c4 = (int)(i & 31) * 4;
    const float4 v = *(const float4*)(h + (long long)t * HD + c4);
    float* ep = emb + (long long)nb[t] * HD + c4;
    atomicAdd(ep + 0, v.x); atomicAdd(ep + 1, v.y);
    atomicAdd(ep + 2, v.z); atomicAdd(ep + 3, v.w);
}
__global__ void k_pool_cnt(const int* __restrict__ nb, float* __restrict__ cnt, int T) {
    int i = blockIdx.x * blockDim.x + threadIdx.x;
    if (i < T) atomicAdd(&cnt[nb[i]], 1.f);
}
__global__ void k_pool_div(float* __restrict__ emb, const float* __restrict__ cnt, int B) {
    long long i = (long long)blockIdx.x * blockDim.x + threadIdx.x;
    if (i >= (long long)B * HD) return;
    emb[i] /= fmaxf(cnt[i >> 7], 1.f);
}

// ---------------- fused MLP row: relu(LN(cat(emb, gf[cid]) @ W + b)) ----------------
__global__ void k_fused(const float* __restrict__ emb, const float* __restrict__ gfeat,
                        const int* __restrict__ cids, const float* __restrict__ W,
                        const float* __restrict__ b, const float* __restrict__ g,
                        const float* __restrict__ beta, float* __restrict__ out, int B) {
    int r = blockIdx.x, c = threadIdx.x;     // 128 threads
    const float* er = emb + (long long)r * HD;
    const float* gr = gfeat + (long long)cids[r] * FGD;
    float acc = b[c];
    for (int k = 0; k < HD; ++k)  acc += er[k] * W[k * HD + c];
    for (int k = 0; k < FGD; ++k) acc += gr[k] * W[(HD + k) * HD + c];
    __shared__ float sh[HD];
    __shared__ float mu, var;
    sh[c] = acc; __syncthreads();
    for (int s = 64; s > 0; s >>= 1) { if (c < s) sh[c] += sh[c + s]; __syncthreads(); }
    if (c == 0) mu = sh[0] * (1.f / HD);
    __syncthreads();
    float dv = acc - mu;
    sh[c] = dv * dv; __syncthreads();
    for (int s = 64; s > 0; s >>= 1) { if (c < s) sh[c] += sh[c + s]; __syncthreads(); }
    if (c == 0) var = sh[0] * (1.f / HD);
    __syncthreads();
    float v = dv * rsqrtf(var + 1e-5f) * g[c] + beta[c];
    out[(long long)r * HD + c] = fmaxf(v, 0.f);
}

// ---------------- global projection row: relu(LN(gf @ W + b)) ----------------
__global__ void k_gproj(const float* __restrict__ gfeat, const float* __restrict__ W,
                        const float* __restrict__ b, const float* __restrict__ g,
                        const float* __restrict__ beta, float* __restrict__ gx, int N) {
    int r = blockIdx.x, c = threadIdx.x;     // 128 threads
    const float* gr = gfeat + (long long)r * FGD;
    float acc = b[c];
#pragma unroll
    for (int k = 0; k < FGD; ++k) acc += gr[k] * W[k * HD + c];
    __shared__ float sh[HD];
    __shared__ float mu, var;
    sh[c] = acc; __syncthreads();
    for (int s = 64; s > 0; s >>= 1) { if (c < s) sh[c] += sh[c + s]; __syncthreads(); }
    if (c == 0) mu = sh[0] * (1.f / HD);
    __syncthreads();
    float dv = acc - mu;
    sh[c] = dv * dv; __syncthreads();
    for (int s = 64; s > 0; s >>= 1) { if (c < s) sh[c] += sh[c + s]; __syncthreads(); }
    if (c == 0) var = sh[0] * (1.f / HD);
    __syncthreads();
    float v = dv * rsqrtf(var + 1e-5f) * g[c] + beta[c];
    gx[(long long)r * HD + c] = fmaxf(v, 0.f);
}

__global__ void k_scatter(const float* __restrict__ fused, const int* __restrict__ cids,
                          float* __restrict__ gx, int B) {
    long long i = (long long)blockIdx.x * blockDim.x + threadIdx.x;
    if (i >= (long long)B * HD) return;
    gx[(long long)cids[i >> 7] * HD + (i & 127)] = fused[i];
}

// ---------------- classifier: out = relu(gx[cid] @ W1 + b1) @ W2 + b2 ----------------
__global__ void k_cls(const float* __restrict__ gx, const int* __restrict__ cids,
                      const float* __restrict__ W1, const float* __restrict__ b1,
                      const float* __restrict__ W2, const float* __restrict__ b2,
                      float* __restrict__ out, int B) {
    int r = blockIdx.x, j = threadIdx.x;     // 64 threads
    __shared__ float hc[HC];
    const float* xr = gx + (long long)cids[r] * HD;
    float acc = b1[j];
    for (int k = 0; k < HD; ++k) acc += xr[k] * W1[k * HC + j];
    hc[j] = fmaxf(acc, 0.f);
    __syncthreads();
    if (j < NCLS) {
        float o = b2[j];
        for (int k = 0; k < HC; ++k) o += hc[k] * W2[k * NCLS + j];
        out[(long long)r * NCLS + j] = o;
    }
}

// =====================================================================
extern "C" void kernel_launch(void* const* d_in, const int* in_sizes, int n_in,
                              void* d_out, int out_size, void* d_ws, size_t ws_size,
                              hipStream_t stream) {
    // inputs (setup_inputs order)
    const float* x_local      = (const float*)d_in[0];
    const float* eattr_local  = (const float*)d_in[1];
    const float* gfeat        = (const float*)d_in[2];
    const float* loc_in_W     = (const float*)d_in[3];
    const float* loc_in_b     = (const float*)d_in[4];
    const float* loc_W        = (const float*)d_in[5];
    const float* loc_asrc     = (const float*)d_in[6];
    const float* loc_adst     = (const float*)d_in[7];
    const float* loc_eW       = (const float*)d_in[8];
    const float* loc_ae       = (const float*)d_in[9];
    const float* loc_b        = (const float*)d_in[10];
    const float* fp_W         = (const float*)d_in[11];
    const float* fp_b         = (const float*)d_in[12];
    const float* fp_g         = (const float*)d_in[13];
    const float* fp_beta      = (const float*)d_in[14];
    const float* gp_W         = (const float*)d_in[15];
    const float* gp_b         = (const float*)d_in[16];
    const float* gp_g         = (const float*)d_in[17];
    const float* gp_beta      = (const float*)d_in[18];
    const float* glob_W       = (const float*)d_in[19];
    const float* glob_asrc    = (const float*)d_in[20];
    const float* glob_adst    = (const float*)d_in[21];
    const float* glob_b       = (const float*)d_in[22];
    const float* cls_W1       = (const float*)d_in[23];
    const float* cls_b1       = (const float*)d_in[24];
    const float* cls_W2       = (const float*)d_in[25];
    const float* cls_b2       = (const float*)d_in[26];
    const int*   eil          = (const int*)d_in[27];   // [2, EL]
    const int*   node_batch   = (const int*)d_in[28];
    const int*   contract_ids = (const int*)d_in[29];
    const int*   gei          = (const int*)d_in[30];   // [2, EG]

    const int T  = in_sizes[0] / NDIM;
    const long long EL = in_sizes[1];
    const int N  = in_sizes[2] / FGD;
    const int B  = in_sizes[29];
    const long long EG = in_sizes[30] / 2;
    const int HLh = 4, HGh = 2;

    // workspace carve-out
    char* ws = (char*)d_ws;
    size_t off = 0;
    auto take = [&](size_t bytes) -> void* {
        void* p = ws + off;
        off += (bytes + 255) & ~(size_t)255;
        return p;
    };
    float*     h    = (float*)take((size_t)T * HD * 4);   // later reused as gx (N rows)
    float*     z    = (float*)take((size_t)T * HD * 4);
    float*     agg  = (float*)take((size_t)T * HD * 4);
    _Float16*  xh   = (_Float16*)take((size_t)T * HD * 2);
    _Float16*  wh   = (_Float16*)take((size_t)HD * HD * 2);
    float*     ss   = (float*)take((size_t)T * 4 * 4);
    float*     sd   = (float*)take((size_t)T * 4 * 4);
    float*     m    = (float*)take((size_t)T * 4 * 4);
    float*     den  = (float*)take((size_t)T * 4 * 4);
    float*     ex   = (float*)take((size_t)EL * 4 * 4);
    float*     emb  = (float*)take((size_t)B * HD * 4);
    float*     cnt  = (float*)take((size_t)B * 4);
    float*     fus  = (float*)take((size_t)B * HD * 4);
    float*     ec   = (float*)take(64 * 4);

    auto G = [](long long n, int b) { return (unsigned)((n + b - 1) / b); };
    const int BL = 256;

    // ---- local input projection ----
    k_in_proj<<<G((long long)T * HD, BL), BL, 0, stream>>>(x_local, loc_in_W, loc_in_b, h, T);

    const int* lsrc = eil;
    const int* ldst = eil + EL;

    // ---- 2x local GAT layers (nh=4, edge attrs) ----
    for (int l = 0; l < 2; ++l) {
        k_cvt_f16<<<G((long long)T * HD, BL), BL, 0, stream>>>(h, xh, (long long)T * HD);
        k_cvt_f16<<<G((long long)HD * HD, BL), BL, 0, stream>>>(loc_W + (size_t)l * HD * HD, wh, (long long)HD * HD);
        k_ec<<<1, 32, 0, stream>>>(loc_eW + (size_t)l * HD, loc_ae + (size_t)l * HD, ec, HLh);
        k_wmma_gemm<<<T / 16, GEMM_THREADS, 0, stream>>>(xh, wh, z, T);
        k_node_scores<<<G((long long)T * HLh, BL), BL, 0, stream>>>(
            z, loc_asrc + (size_t)l * HD, loc_adst + (size_t)l * HD, ss, sd, T, HLh);
        k_fill<<<G((long long)T * HLh, BL), BL, 0, stream>>>(m, -__builtin_inff(), (long long)T * HLh);
        k_fill<<<G((long long)T * HLh, BL), BL, 0, stream>>>(den, 0.f, (long long)T * HLh);
        k_fill<<<G((long long)T * HD, BL), BL, 0, stream>>>(agg, 0.f, (long long)T * HD);
        k_edge_score<<<G(EL * HLh, BL), BL, 0, stream>>>(lsrc, ldst, ss, sd, eattr_local, ec, ex, m, EL, HLh);
        k_m_fix<<<G((long long)T * HLh, BL), BL, 0, stream>>>(m, (long long)T * HLh);
        k_edge_exp<<<G(EL * HLh, BL), BL, 0, stream>>>(ldst, m, ex, den, EL, HLh);
        k_edge_agg<<<G(EL * 32, BL), BL, 0, stream>>>(lsrc, ldst, z, ex, den, agg, EL, HLh);
        k_res_elu<<<G((long long)T * HD, BL), BL, 0, stream>>>(h, agg, loc_b + (size_t)l * HD, T);
    }

    // ---- contract mean pooling ----
    k_fill<<<G((long long)B * HD, BL), BL, 0, stream>>>(emb, 0.f, (long long)B * HD);
    k_fill<<<G(B, BL), BL, 0, stream>>>(cnt, 0.f, B);
    k_pool_sum<<<G((long long)T * 32, BL), BL, 0, stream>>>(h, node_batch, emb, T);
    k_pool_cnt<<<G(T, BL), BL, 0, stream>>>(node_batch, cnt, T);
    k_pool_div<<<G((long long)B * HD, BL), BL, 0, stream>>>(emb, cnt, B);

    // ---- fused MLP + global projection + scatter (gx aliases h; h is dead now) ----
    k_fused<<<B, HD, 0, stream>>>(emb, gfeat, contract_ids, fp_W, fp_b, fp_g, fp_beta, fus, B);
    float* gx = h;
    k_gproj<<<N, HD, 0, stream>>>(gfeat, gp_W, gp_b, gp_g, gp_beta, gx, N);
    k_scatter<<<G((long long)B * HD, BL), BL, 0, stream>>>(fus, contract_ids, gx, B);

    const int* gsrc = gei;
    const int* gdst = gei + EG;

    // ---- 2x global GAT layers (nh=2, no edge attrs) ----
    for (int l = 0; l < 2; ++l) {
        k_cvt_f16<<<G((long long)N * HD, BL), BL, 0, stream>>>(gx, xh, (long long)N * HD);
        k_cvt_f16<<<G((long long)HD * HD, BL), BL, 0, stream>>>(glob_W + (size_t)l * HD * HD, wh, (long long)HD * HD);
        k_wmma_gemm<<<N / 16, GEMM_THREADS, 0, stream>>>(xh, wh, z, N);
        k_node_scores<<<G((long long)N * HGh, BL), BL, 0, stream>>>(
            z, glob_asrc + (size_t)l * HD, glob_adst + (size_t)l * HD, ss, sd, N, HGh);
        k_fill<<<G((long long)N * HGh, BL), BL, 0, stream>>>(m, -__builtin_inff(), (long long)N * HGh);
        k_fill<<<G((long long)N * HGh, BL), BL, 0, stream>>>(den, 0.f, (long long)N * HGh);
        k_fill<<<G((long long)N * HD, BL), BL, 0, stream>>>(agg, 0.f, (long long)N * HD);
        k_edge_score<<<G(EG * HGh, BL), BL, 0, stream>>>(gsrc, gdst, ss, sd, nullptr, ec, ex, m, EG, HGh);
        k_m_fix<<<G((long long)N * HGh, BL), BL, 0, stream>>>(m, (long long)N * HGh);
        k_edge_exp<<<G(EG * HGh, BL), BL, 0, stream>>>(gdst, m, ex, den, EG, HGh);
        k_edge_agg<<<G(EG * 32, BL), BL, 0, stream>>>(gsrc, gdst, z, ex, den, agg, EG, HGh);
        k_res_elu<<<G((long long)N * HD, BL), BL, 0, stream>>>(gx, agg, glob_b + (size_t)l * HD, N);
    }

    // ---- classifier ----
    k_cls<<<B, HC, 0, stream>>>(gx, contract_ids, cls_W1, cls_b1, cls_W2, cls_b2, (float*)d_out, B);
}
